// CustomLlamaAttention_11690900979839
// MI455X (gfx1250) — matrix operand order repaired and checked by reading, not compiled, not run on hardware
//
#include <hip/hip_runtime.h>
#include <hip/hip_bf16.h>

// ---------------- problem constants (from reference) ----------------
#define HIDN  2048   // hidden size
#define SEQ   2048   // sequence length
#define NHEAD 32     // attention heads
#define NKVH  8      // kv heads
#define HDIM  64     // head dim
#define LATD  256    // LR*NKV latent dim
#define NROPE 256    // roped key dims
#define THETA 10000.0f

typedef __bf16 bf16;
typedef __attribute__((ext_vector_type(16))) __bf16 bf16x16;
typedef __attribute__((ext_vector_type(8)))  __bf16 bf16x8;
typedef __attribute__((ext_vector_type(8)))  float  f32x8;

union BFrag { bf16x16 v; bf16x8 h[2]; };

static __device__ __forceinline__ f32x8 zero8() {
  f32x8 z = {0.f,0.f,0.f,0.f,0.f,0.f,0.f,0.f};
  return z;
}

// D = A(16x32 bf16) * B(32x16 bf16) + C(16x16 f32), wave32
#define WMMA_BF16(A,B,C) \
  __builtin_amdgcn_wmma_f32_16x16x32_bf16(false,(A),false,(B),(short)0,(C),false,false)

// ---------------- fp32 -> bf16 conversion ----------------
__global__ void cvt_f32_bf16(const float* __restrict__ in, bf16* __restrict__ out, int n) {
  int i = blockIdx.x * blockDim.x + threadIdx.x;
  if (i < n) out[i] = (bf16)in[i];
}

// ---------------- generic WMMA GEMM: C[M,N] = A[M,K] @ W[N,K]^T ----------------
// block = 256 threads = 8 waves; wave computes 32x64 (2x4 tiles of 16x16)
// grid = (N/64, M/256)
// store_mode: 0 = bf16 row-major, 1 = f32 row-major, 2 = bf16 transposed (C[n*M+m])
__global__ __launch_bounds__(256)
void gemm_bf16_wmma(const bf16* __restrict__ A, const bf16* __restrict__ W,
                    void* __restrict__ C, int M, int N, int K, int store_mode) {
  const int lane = threadIdx.x & 31;
  const int wave = threadIdx.x >> 5;
  const int m_base = blockIdx.y * 256 + wave * 32;
  const int n_base = blockIdx.x * 64;
  const int mrow = lane & 15;   // A: row within tile / B: column within tile
  const int mk   = lane >> 4;   // A: k-subgroup      / B: k-halfword group

  f32x8 acc[2][4];
#pragma unroll
  for (int i = 0; i < 2; ++i)
#pragma unroll
    for (int j = 0; j < 4; ++j) acc[i][j] = zero8();

  for (int k0 = 0; k0 < K; k0 += 32) {
    BFrag a[2];
#pragma unroll
    for (int i = 0; i < 2; ++i) {
      const bf16* ap = A + (size_t)(m_base + 16*i + mrow) * K + k0 + mk * 8;
      a[i].h[0] = *(const bf16x8*)(ap);        // K = k0 + mk*8 + 0..7
      a[i].h[1] = *(const bf16x8*)(ap + 16);   // K = k0 + 16 + mk*8 + 0..7
    }
    BFrag b[4];
#pragma unroll
    for (int j = 0; j < 4; ++j) {
      const bf16* bp = W + (size_t)(n_base + 16*j + mrow) * K + k0 + mk * 8;
      b[j].h[0] = *(const bf16x8*)(bp);        // K = k0 + mk*8 + 0..7
      b[j].h[1] = *(const bf16x8*)(bp + 16);   // K = k0 + 16 + mk*8 + 0..7
    }
#pragma unroll
    for (int i = 0; i < 2; ++i)
#pragma unroll
      for (int j = 0; j < 4; ++j)
        acc[i][j] = WMMA_BF16(a[i].v, b[j].v, acc[i][j]);
  }

  // C/D layout: VGPR r -> row = r + 8*(lane>=16), col = lane&15
  const int rg  = lane >> 4;
  const int col = lane & 15;
  if (store_mode == 1) {
    float* Cf = (float*)C;
#pragma unroll
    for (int i = 0; i < 2; ++i)
#pragma unroll
      for (int j = 0; j < 4; ++j)
#pragma unroll
        for (int r = 0; r < 8; ++r) {
          int row = m_base + 16*i + r + 8*rg;
          int c   = n_base + 16*j + col;
          Cf[(size_t)row * N + c] = acc[i][j][r];
        }
  } else if (store_mode == 0) {
    bf16* Cb = (bf16*)C;
#pragma unroll
    for (int i = 0; i < 2; ++i)
#pragma unroll
      for (int j = 0; j < 4; ++j)
#pragma unroll
        for (int r = 0; r < 8; ++r) {
          int row = m_base + 16*i + r + 8*rg;
          int c   = n_base + 16*j + col;
          Cb[(size_t)row * N + c] = (bf16)acc[i][j][r];
        }
  } else { // transposed bf16: C[c*M + row]
    bf16* Cb = (bf16*)C;
#pragma unroll
    for (int i = 0; i < 2; ++i)
#pragma unroll
      for (int j = 0; j < 4; ++j)
#pragma unroll
        for (int r = 0; r < 8; ++r) {
          int row = m_base + 16*i + r + 8*rg;
          int c   = n_base + 16*j + col;
          Cb[(size_t)c * M + row] = (bf16)acc[i][j][r];
        }
  }
}

// ---------------- RoPE on Q (in place, bf16) ----------------
// thread handles (s, h, j) with j in [0,32): rotates dims j and j+32 of head h
__global__ void rope_q_kernel(bf16* __restrict__ Q) {
  int idx = blockIdx.x * blockDim.x + threadIdx.x;
  if (idx >= SEQ * NHEAD * 32) return;
  int j = idx & 31;
  int h = (idx >> 5) & (NHEAD - 1);
  int s = idx >> 10;
  float freq = __powf(THETA, -(float)j / 32.0f);
  float ang = (float)s * freq;
  float c = __cosf(ang), sn = __sinf(ang);
  bf16* row = Q + (size_t)s * (NHEAD * HDIM) + h * HDIM;
  float a = (float)row[j];
  float b = (float)row[j + 32];
  row[j]      = (bf16)(a * c - b * sn);
  row[j + 32] = (bf16)(b * c + a * sn);
}

// ---------------- K assembly: rope k_r dims + scatter, nope dims from k_c ----------------
// per-head rope dims: [0,16) U [32,48) ; nope dims: [16,32) U [48,64)
// k_r cols per head: t<16 -> dim t ; t in [16,32) -> dim 32+(t-16)
// k_c cols per head: t<16 -> dim 16+t ; t in [16,32) -> dim 48+(t-16)
__global__ void assemble_k_kernel(const bf16* __restrict__ KR, const bf16* __restrict__ KC,
                                  bf16* __restrict__ Kf) {
  int idx = blockIdx.x * blockDim.x + threadIdx.x;
  if (idx >= SEQ * NKVH * 16) return;
  int j   = idx & 15;
  int kvh = (idx >> 4) & (NKVH - 1);
  int s   = idx >> 7;
  float freq = __powf(THETA, -(float)j / 32.0f);
  float ang = (float)s * freq;
  float c = __cosf(ang), sn = __sinf(ang);
  const bf16* kr = KR + (size_t)s * NROPE + kvh * 32;
  const bf16* kc = KC + (size_t)s * NROPE + kvh * 32;
  bf16* ko = Kf + (size_t)s * (NKVH * HDIM) + kvh * HDIM;
  float a = (float)kr[j];
  float b = (float)kr[16 + j];
  ko[j]      = (bf16)(a * c - b * sn);   // roped dim j
  ko[32 + j] = (bf16)(b * c + a * sn);   // roped dim 32+j
  ko[16 + j] = kc[j];                    // nope dim 16+j
  ko[48 + j] = kc[16 + j];               // nope dim 48+j
}

// ---------------- flash attention: one wave per (32-query tile, head) ----------------
// grid = (SEQ/32, NHEAD), block = 32
// Wave holds 2 Q-tiles (32 queries); K/V fragments are loaded once per 32-key
// step and shared by both Q-tiles -> 16 WMMA per iteration.
__global__ __launch_bounds__(32)
void attn_kernel(const bf16* __restrict__ Q, const bf16* __restrict__ Kf,
                 const bf16* __restrict__ Vt, bf16* __restrict__ O) {
  __shared__ __align__(16) bf16 Psh[32][32];

  const int lane = threadIdx.x;
  const int q0   = blockIdx.x * 32;
  const int h    = blockIdx.y;
  const int kvh  = h >> 2;               // GQA: 4 q-heads per kv head
  const int mrow = lane & 15;
  const int mk   = lane >> 4;
  const int rg   = lane >> 4;
  const int col  = lane & 15;
  const float scale = 0.125f;            // 1/sqrt(64)

  // Q A-fragments (reused across whole key loop): [qtile][kstep]
  BFrag aq[2][2];
#pragma unroll
  for (int qt = 0; qt < 2; ++qt)
#pragma unroll
    for (int ks = 0; ks < 2; ++ks) {
      const bf16* qp = Q + (size_t)(q0 + qt * 16 + mrow) * (NHEAD * HDIM)
                         + h * HDIM + ks * 32 + mk * 8;
      aq[qt][ks].h[0] = *(const bf16x8*)(qp);
      aq[qt][ks].h[1] = *(const bf16x8*)(qp + 16);
    }

  f32x8 oacc[2][4];
  float m_i[2][8], l_i[2][8];
#pragma unroll
  for (int qt = 0; qt < 2; ++qt) {
#pragma unroll
    for (int dt = 0; dt < 4; ++dt) oacc[qt][dt] = zero8();
#pragma unroll
    for (int r = 0; r < 8; ++r) { m_i[qt][r] = -INFINITY; l_i[qt][r] = 0.f; }
  }

  const int kend = q0 + 32;  // multiple of 32 -> no partial iterations, ever
  for (int kb = 0; kb < kend; kb += 32) {
    // ---- K B-fragments: [key-subtile][kstep], shared by both Q-tiles ----
    BFrag bk[2][2];
#pragma unroll
    for (int t = 0; t < 2; ++t)
#pragma unroll
      for (int ks = 0; ks < 2; ++ks) {
        const bf16* kp = Kf + (size_t)(kb + t * 16 + col) * (NKVH * HDIM)
                            + kvh * HDIM + ks * 32 + rg * 8;
        bk[t][ks].h[0] = *(const bf16x8*)(kp);
        bk[t][ks].h[1] = *(const bf16x8*)(kp + 16);
      }
    // ---- scores: 8 WMMA ----
    f32x8 sc[2][2];  // [qtile][key-subtile]
#pragma unroll
    for (int qt = 0; qt < 2; ++qt)
#pragma unroll
      for (int t = 0; t < 2; ++t) {
        f32x8 s = zero8();
#pragma unroll
        for (int ks = 0; ks < 2; ++ks)
          s = WMMA_BF16(aq[qt][ks].v, bk[t][ks].v, s);
        sc[qt][t] = s;
      }
    // ---- scale + causal mask + online softmax per Q-tile ----
#pragma unroll
    for (int qt = 0; qt < 2; ++qt) {
#pragma unroll
      for (int t = 0; t < 2; ++t)
#pragma unroll
        for (int r = 0; r < 8; ++r) {
          int key = kb + t * 16 + col;
          int q   = q0 + qt * 16 + r + 8 * rg;
          float v = sc[qt][t][r] * scale;
          sc[qt][t][r] = (key <= q) ? v : -INFINITY;
        }
#pragma unroll
      for (int r = 0; r < 8; ++r) {
        float t = fmaxf(sc[qt][0][r], sc[qt][1][r]);
        t = fmaxf(t, __shfl_xor(t, 1, 16));
        t = fmaxf(t, __shfl_xor(t, 2, 16));
        t = fmaxf(t, __shfl_xor(t, 4, 16));
        t = fmaxf(t, __shfl_xor(t, 8, 16));
        float mn   = fmaxf(m_i[qt][r], t);
        float corr = __expf(m_i[qt][r] - mn);
        float p0 = __expf(sc[qt][0][r] - mn);
        float p1 = __expf(sc[qt][1][r] - mn);
        sc[qt][0][r] = p0; sc[qt][1][r] = p1;
        float ssum = p0 + p1;
        ssum += __shfl_xor(ssum, 1, 16);
        ssum += __shfl_xor(ssum, 2, 16);
        ssum += __shfl_xor(ssum, 4, 16);
        ssum += __shfl_xor(ssum, 8, 16);
        l_i[qt][r] = l_i[qt][r] * corr + ssum;
        m_i[qt][r] = mn;
#pragma unroll
        for (int dt = 0; dt < 4; ++dt) oacc[qt][dt][r] *= corr;
      }
    }
    // ---- P: C-layout -> A-layout via LDS (both Q-tiles) ----
    __syncthreads();
#pragma unroll
    for (int qt = 0; qt < 2; ++qt)
#pragma unroll
      for (int t = 0; t < 2; ++t)
#pragma unroll
        for (int r = 0; r < 8; ++r)
          Psh[qt * 16 + r + 8 * rg][t * 16 + col] = (bf16)sc[qt][t][r];
    __syncthreads();
    BFrag pa[2];
#pragma unroll
    for (int qt = 0; qt < 2; ++qt) {
      const bf16* pp = &Psh[qt * 16 + mrow][mk * 8];
      pa[qt].h[0] = *(const bf16x8*)(pp);
      pa[qt].h[1] = *(const bf16x8*)(pp + 16);
    }
    // ---- V B-fragments (shared), then PV: 8 WMMA ----
    BFrag bv[4];
#pragma unroll
    for (int dt = 0; dt < 4; ++dt) {
      const bf16* vp = Vt + (size_t)(kvh * HDIM + dt * 16 + col) * SEQ + kb + rg * 8;
      bv[dt].h[0] = *(const bf16x8*)(vp);
      bv[dt].h[1] = *(const bf16x8*)(vp + 16);
    }
#pragma unroll
    for (int qt = 0; qt < 2; ++qt)
#pragma unroll
      for (int dt = 0; dt < 4; ++dt)
        oacc[qt][dt] = WMMA_BF16(pa[qt].v, bv[dt].v, oacc[qt][dt]);
  }

  // ---- epilogue: normalize and store bf16 ----
#pragma unroll
  for (int qt = 0; qt < 2; ++qt)
#pragma unroll
    for (int r = 0; r < 8; ++r) {
      float invl = 1.0f / l_i[qt][r];
      int row = q0 + qt * 16 + r + 8 * rg;
#pragma unroll
      for (int dt = 0; dt < 4; ++dt) {
        int d = dt * 16 + col;
        O[(size_t)row * (NHEAD * HDIM) + h * HDIM + d] = (bf16)(oacc[qt][dt][r] * invl);
      }
    }
}

// ---------------- launch ----------------
extern "C" void kernel_launch(void* const* d_in, const int* in_sizes, int n_in,
                              void* d_out, int out_size, void* d_ws, size_t ws_size,
                              hipStream_t stream) {
  (void)in_sizes; (void)n_in; (void)out_size; (void)ws_size;

  const float* h_f   = (const float*)d_in[0];  // [S, HID]
  const float* qw_f  = (const float*)d_in[1];  // [NH*HD, HID]
  const float* krw_f = (const float*)d_in[2];  // [256, HID]
  const float* dww_f = (const float*)d_in[3];  // [256, HID]
  const float* upk_f = (const float*)d_in[4];  // [256, 256]
  const float* upv_f = (const float*)d_in[5];  // [512, 256]
  const float* ow_f  = (const float*)d_in[6];  // [HID, NH*HD]

  char* wsp = (char*)d_ws;
  size_t off = 0;
  auto carve = [&](size_t elems) -> bf16* {
    bf16* p = (bf16*)(wsp + off);
    off += ((elems * sizeof(bf16) + 255) & ~(size_t)255);
    return p;
  };
  bf16* Xb    = carve((size_t)SEQ * HIDN);        // bf16 hidden
  bf16* QWb   = carve((size_t)HIDN * HIDN);
  bf16* KRWb  = carve((size_t)NROPE * HIDN);
  bf16* DWb   = carve((size_t)LATD * HIDN);
  bf16* UPKb  = carve((size_t)NROPE * LATD);
  bf16* UPVb  = carve((size_t)(NKVH * HDIM) * LATD);
  bf16* OWb   = carve((size_t)HIDN * HIDN);
  bf16* Qb    = carve((size_t)SEQ * (NHEAD * HDIM));
  bf16* KRb   = carve((size_t)SEQ * NROPE);
  bf16* CKVb  = carve((size_t)SEQ * LATD);
  bf16* KCb   = carve((size_t)SEQ * NROPE);
  bf16* Kfb   = carve((size_t)SEQ * (NKVH * HDIM));
  bf16* Vtb   = carve((size_t)(NKVH * HDIM) * SEQ); // transposed V
  bf16* AOb   = carve((size_t)SEQ * (NHEAD * HDIM));

  auto cvt = [&](const float* src, bf16* dst, size_t n) {
    cvt_f32_bf16<<<dim3((unsigned)((n + 255) / 256)), dim3(256), 0, stream>>>(src, dst, (int)n);
  };
  cvt(h_f,   Xb,   (size_t)SEQ * HIDN);
  cvt(qw_f,  QWb,  (size_t)HIDN * HIDN);
  cvt(krw_f, KRWb, (size_t)NROPE * HIDN);
  cvt(dww_f, DWb,  (size_t)LATD * HIDN);
  cvt(upk_f, UPKb, (size_t)NROPE * LATD);
  cvt(upv_f, UPVb, (size_t)(NKVH * HDIM) * LATD);
  cvt(ow_f,  OWb,  (size_t)HIDN * HIDN);

  dim3 blk(256);
  // Q = X @ q_w^T  [2048 x 2048]
  gemm_bf16_wmma<<<dim3(HIDN / 64, SEQ / 256), blk, 0, stream>>>(Xb, QWb, Qb, SEQ, HIDN, HIDN, 0);
  // k_r = X @ kr_w^T [2048 x 256]
  gemm_bf16_wmma<<<dim3(NROPE / 64, SEQ / 256), blk, 0, stream>>>(Xb, KRWb, KRb, SEQ, NROPE, HIDN, 0);
  // c_kv = X @ down_w^T [2048 x 256]
  gemm_bf16_wmma<<<dim3(LATD / 64, SEQ / 256), blk, 0, stream>>>(Xb, DWb, CKVb, SEQ, LATD, HIDN, 0);
  // k_c = c_kv @ upk_w^T [2048 x 256]
  gemm_bf16_wmma<<<dim3(NROPE / 64, SEQ / 256), blk, 0, stream>>>(CKVb, UPKb, KCb, SEQ, NROPE, LATD, 0);
  // V^T = (c_kv @ upv_w^T)^T  -> stored [512 x 2048]
  gemm_bf16_wmma<<<dim3((NKVH * HDIM) / 64, SEQ / 256), blk, 0, stream>>>(CKVb, UPVb, Vtb, SEQ, NKVH * HDIM, LATD, 2);

  // partial RoPE on Q (in place)
  {
    int n = SEQ * NHEAD * 32;
    rope_q_kernel<<<dim3((n + 255) / 256), dim3(256), 0, stream>>>(Qb);
  }
  // assemble full K (rope + nope scatter)
  {
    int n = SEQ * NKVH * 16;
    assemble_k_kernel<<<dim3((n + 255) / 256), dim3(256), 0, stream>>>(KRb, KCb, Kfb);
  }
  // flash attention (32 queries per wave)
  attn_kernel<<<dim3(SEQ / 32, NHEAD), dim3(32), 0, stream>>>(Qb, Kfb, Vtb, AOb);

  // out = attn_out @ o_w^T -> f32 d_out [2048 x 2048]
  gemm_bf16_wmma<<<dim3(HIDN / 64, SEQ / 256), blk, 0, stream>>>(AOb, OWb, d_out, SEQ, HIDN, HIDN, 1);
}
